// GRU_ODE_69681549410698
// MI455X (gfx1250) — matrix operand order repaired and checked by reading, not compiled, not run on hardware
//
#include <hip/hip_runtime.h>
#include <hip/hip_bf16.h>
#include <math.h>
#include <stdint.h>

// Problem dims
#define B_   128
#define T_   256
#define E_   256
#define Z_   256
#define U_   128
#define H_   128
#define G3_  768   // 3*Z
#define GH_  384   // 3*H
#define KP_  288   // 257 padded up to a multiple of 32

// LDS pitches (elements): +8 bf16 (16B) pad per row -> 4-bank lane stride,
// conflict-free B-fragment reads.  Matches TDM pad_amount = 4 DWORDs.
#define PITCH_GRU   (Z_ + 8)   // 264 elems = 528B rows; 128 rows = 67584B
#define PITCH_BIGRU (H_ + 8)   // 136 elems = 272B rows; 128 rows = 34816B

typedef __bf16 bf16;
typedef __attribute__((ext_vector_type(16))) __bf16 v16bf;
typedef __attribute__((ext_vector_type(8)))  __bf16 v8bf;
typedef __attribute__((ext_vector_type(8)))  float   v8f;
typedef __attribute__((ext_vector_type(4)))  unsigned v4u;
typedef __attribute__((ext_vector_type(8)))  int      v8i;
typedef __attribute__((ext_vector_type(4)))  int      v4i;

#if __has_builtin(__builtin_amdgcn_tensor_load_to_lds) && \
    __has_builtin(__builtin_amdgcn_s_wait_tensorcnt)
#define USE_TDM 1
#else
#define USE_TDM 0
#endif

#define DEVI static __device__ __forceinline__

DEVI bf16 f2bf(float f) {
  unsigned u = __float_as_uint(f);
  unsigned r = u + 0x7FFFu + ((u >> 16) & 1u);   // round-to-nearest-even
  unsigned short h = (unsigned short)(r >> 16);
  return __builtin_bit_cast(bf16, h);
}
DEVI float sigm(float x) { return 1.0f / (1.0f + __expf(-x)); }
DEVI int imax0(int a) { return a > 0 ? a : 0; }

// ---------------------------------------------------------------------------
// WMMA fragment helpers (wave32; this code's contiguous k order, same in A and B):
//  A frag: lane l holds row m0+(l&15), K chunk  [k0+(l>>4)*16..+16).
//  B frag: lane l holds col n0+(l&15), K chunk  [k0+(l>>4)*16..+16) -> one
//          contiguous run of W's row (W stored N x K row-major).
//  D elem r -> (m0 + (l>>4)*8 + r, n0 + (l&15)).
// ---------------------------------------------------------------------------
DEVI v16bf pack16(v8bf lo, v8bf hi8) {
  v16bf v;
#pragma unroll
  for (int i = 0; i < 8; ++i) { v[i] = lo[i]; v[i + 8] = hi8[i]; }
  return v;
}

// Single-tile K-loop, global A and global W (used by small recurrent GEMMs).
DEVI v8f wmma_tile(const bf16* __restrict__ A, int lda,
                   const bf16* __restrict__ W, int ldw,
                   int m0, int n0, int K, int lane, v8f acc) {
  const int l = lane & 15, hi = lane >> 4;
  const bf16* pa = A + (size_t)(m0 + l) * lda + hi * 16;
  const bf16* pb = W + (size_t)(n0 + l) * ldw + hi * 16;
  for (int k = 0; k < K; k += 32) {
    v8bf a0 = *(const v8bf*)(pa + k);
    v8bf a1 = *(const v8bf*)(pa + k + 8);
    v8bf b0 = *(const v8bf*)(pb + k);
    v8bf b1 = *(const v8bf*)(pb + k + 8);
    __builtin_prefetch(pa + k + 256, 0, 1);   // global_prefetch_b8
    v16bf a = pack16(a0, a1), b = pack16(b0, b1);
    acc = __builtin_amdgcn_wmma_f32_16x16x32_bf16(false, a, false, b,
                                                  (short)0, acc, false, false);
  }
  return acc;
}

// Single-tile K-loop with B in LDS (row-major, padded pitch).
DEVI v8f wmma_tile_ldsB(const bf16* __restrict__ A, int lda,
                        const bf16* ldsW, int pitch, int nl0,
                        int m0, int K, int lane, v8f acc) {
  const int l = lane & 15, hi = lane >> 4;
  const bf16* pa  = A + (size_t)(m0 + l) * lda + hi * 16;
  const bf16* pb0 = ldsW + (size_t)(nl0 + l) * pitch + hi * 16;
  for (int k = 0; k < K; k += 32) {
    v8bf a0 = *(const v8bf*)(pa + k);
    v8bf a1 = *(const v8bf*)(pa + k + 8);
    v8bf b0 = *(const v8bf*)(pb0 + k);
    v8bf b1 = *(const v8bf*)(pb0 + k + 8);
    v16bf a = pack16(a0, a1), b = pack16(b0, b1);
    acc = __builtin_amdgcn_wmma_f32_16x16x32_bf16(false, a, false, b,
                                                  (short)0, acc, false, false);
  }
  return acc;
}

// ---------------------------------------------------------------------------
// TDM staging: DMA a rows x cols bf16 tile (row stride == cols) into LDS at
// lds_off, padding 4 DWORDs (16B) after each row (pad_interval selects a
// power-of-two row length in 8B units: (1<<code)*8 bytes == cols*2).
// D# layout per CDNA5 ISA 8.3/8.4 (2D tensor, groups 2/3 zero).
// Must be called by ONE wave per block (wave-uniform args); tracked with
// TENSORcnt on the issuing wave; others synchronize via __syncthreads().
// ---------------------------------------------------------------------------
#if USE_TDM
DEVI void stage_tile_tdm(const bf16* gsrc, unsigned lds_off,
                         unsigned rows, unsigned cols, unsigned pad_code) {
  const unsigned long long ga = (unsigned long long)(uintptr_t)gsrc;
  v4u g0;
  g0[0] = 1u;                                        // count=1, user mode
  g0[1] = lds_off;                                   // lds_addr [63:32]
  g0[2] = (unsigned)(ga & 0xFFFFFFFFu);              // global_addr lo
  g0[3] = (unsigned)((ga >> 32) & 0x01FFFFFFu) | 0x80000000u;  // hi | type=2
  v8i g1;
  g1[0] = (int)((1u << 16)               // data_size = 2B
              | (1u << 20)               // pad_enable
              | (pad_code << 22)         // pad_interval
              | (3u << 25));             // pad_amount = 4 DWORDs
  g1[1] = (int)((cols & 0xFFFFu) << 16);             // tensor_dim0 [79:48] lo
  g1[2] = (int)((cols >> 16) | ((rows & 0xFFFFu) << 16)); // dim0 hi | dim1 lo
  g1[3] = (int)((rows >> 16) | ((cols & 0xFFFFu) << 16)); // dim1 hi | tile_dim0
  g1[4] = (int)(rows & 0xFFFFu);                     // tile_dim1 (tile_dim2=0)
  g1[5] = (int)cols;                                 // tensor_dim0_stride lo
  g1[6] = 0;                                         // stride hi | dim1_stride
  g1[7] = 0;
  v4i gz4 = {0, 0, 0, 0};
  v8i gz8 = {0, 0, 0, 0, 0, 0, 0, 0};
  __builtin_amdgcn_tensor_load_to_lds(g0, g1, gz4, gz4, gz8, 0);
}
#endif

// Fallback / portable staging into the identical padded row-major layout.
DEVI void stage_tile_manual(const bf16* __restrict__ src, bf16* ldsW,
                            int rows, int cols, int pitch) {
  const int chunks = cols / 8;
  for (int i = threadIdx.x; i < rows * chunks; i += blockDim.x) {
    const int r = i / chunks, c = (i % chunks) * 8;
    *(v8bf*)(ldsW + (size_t)r * pitch + c) =
        *(const v8bf*)(src + (size_t)r * cols + c);
  }
}

// Stage a weight slice into LDS; one TDM op (preferred) or manual copy.
DEVI void stage_slice(const bf16* gsrc, bf16* ldsW, int rows, int cols,
                      int pitch, unsigned pad_code) {
#if USE_TDM
  (void)pitch;
  if (threadIdx.x < 32) {
    stage_tile_tdm(gsrc, (unsigned)(uintptr_t)ldsW, rows, cols, pad_code);
    __builtin_amdgcn_s_wait_tensorcnt(0);
  }
#else
  (void)pad_code;
  stage_tile_manual(gsrc, ldsW, rows, cols, pitch);
#endif
}

// Sense-reversing grid barrier over all blocks of the launch.
DEVI void grid_sync(unsigned* bar) {
  __threadfence();
  __syncthreads();
  if (threadIdx.x == 0) {
    unsigned gen = __hip_atomic_load(bar + 1, __ATOMIC_RELAXED, __HIP_MEMORY_SCOPE_AGENT);
    unsigned arr = __hip_atomic_fetch_add(bar, 1u, __ATOMIC_ACQ_REL, __HIP_MEMORY_SCOPE_AGENT) + 1u;
    if (arr == gridDim.x) {
      __hip_atomic_store(bar, 0u, __ATOMIC_RELAXED, __HIP_MEMORY_SCOPE_AGENT);
      __hip_atomic_fetch_add(bar + 1, 1u, __ATOMIC_RELEASE, __HIP_MEMORY_SCOPE_AGENT);
    } else {
      while (__hip_atomic_load(bar + 1, __ATOMIC_ACQUIRE, __HIP_MEMORY_SCOPE_AGENT) == gen)
        __builtin_amdgcn_s_sleep(1);
    }
  }
  __syncthreads();
}

// ---------------------------------------------------------------------------
// Weight conversion helpers
// ---------------------------------------------------------------------------
__global__ void cvt_bf16(const float* __restrict__ src, bf16* __restrict__ dst, int n) {
  int i = blockIdx.x * blockDim.x + threadIdx.x;
  if (i < n) dst[i] = f2bf(src[i]);
}

__global__ void pad_wo1(const float* __restrict__ src, bf16* __restrict__ dst) {
  // (U x 257) f32 -> (U x 288) bf16, zero padded
  int i = blockIdx.x * blockDim.x + threadIdx.x;
  if (i < U_ * KP_) {
    int r = i / KP_, c = i % KP_;
    dst[i] = f2bf(c < (Z_ + 1) ? src[r * (Z_ + 1) + c] : 0.0f);
  }
}

// ---------------------------------------------------------------------------
// Stage 1: temporal conv + enhancement -> enh (bf16 GEMM A operand).
// ---------------------------------------------------------------------------
__global__ __launch_bounds__(256) void tconv_kernel(
    const float* __restrict__ x, const float* __restrict__ ts,
    const float* __restrict__ Wc1, const float* __restrict__ bc1,
    const float* __restrict__ Wc2, const float* __restrict__ bc2,
    const float* __restrict__ Wc3, const float* __restrict__ bc3,
    bf16* __restrict__ enh_b) {
  const int bt = blockIdx.x;
  const int b = bt / T_, t = bt % T_;
  __shared__ float wsm[4];
  const int tid = threadIdx.x;
  if (tid < 4) {
    const int d = tid + 1;
    float w = 0.0f;
    if (t >= d) {                                    // valid mask folded into w
      float tdv = fabsf(ts[b * T_ + t] - ts[b * T_ + (t - d)]);
      tdv = fmaxf(tdv, 1e-6f);
      const float f0 = tdv;
      const float f1 = logf(tdv + 1e-6f);
      const float cws = fminf((float)(t + 1), 5.0f);
      const float f2 = (float)d / cws;
      float h1[32];
#pragma unroll
      for (int j = 0; j < 32; ++j)
        h1[j] = fmaxf(Wc1[j * 3 + 0] * f0 + Wc1[j * 3 + 1] * f1 +
                      Wc1[j * 3 + 2] * f2 + bc1[j], 0.0f);
      float acc3 = bc3[0];
#pragma unroll
      for (int j = 0; j < 16; ++j) {
        float sj = bc2[j];
#pragma unroll
        for (int i = 0; i < 32; ++i) sj += Wc2[j * 32 + i] * h1[i];
        acc3 += Wc3[j] * fmaxf(sj, 0.0f);
      }
      w = sigm(acc3);
    }
    wsm[tid] = w;
  }
  __syncthreads();
  const float w1 = wsm[0], w2 = wsm[1], w3 = wsm[2], w4 = wsm[3];
  const float den = fmaxf(w1 + w2 + w3 + w4 + 1.0f, 1.0f);
  const int e = tid;
  const size_t base = (size_t)b * T_ * E_;
  const float xv = x[base + (size_t)t * E_ + e];
  float num = xv;
  num += w1 * x[base + (size_t)imax0(t - 1) * E_ + e];
  num += w2 * x[base + (size_t)imax0(t - 2) * E_ + e];
  num += w3 * x[base + (size_t)imax0(t - 3) * E_ + e];
  num += w4 * x[base + (size_t)imax0(t - 4) * E_ + e];
  enh_b[(size_t)bt * E_ + e] = f2bf(xv + num / den);
}

// ---------------------------------------------------------------------------
// Generic big WMMA GEMM, register-blocked 1x4: each wave computes a 16x64
// output strip (one A fragment feeds 4 WMMAs) -> 4x less A traffic.
// C[M,N] = A_bf16[M,KC] @ W_bf16[N,KC]^T + bias.
// permute!=0 remaps output row m = t*PB + b  ->  (b*PT + t).
// ---------------------------------------------------------------------------
template <int KC>
__global__ __launch_bounds__(256) void gemm_bias_kernel(
    const bf16* __restrict__ A, int lda,
    const bf16* __restrict__ W, int ldw,
    const float* __restrict__ bias, float* __restrict__ C,
    int M, int N, int permute, int PB, int PT) {
  const int lane = threadIdx.x & 31;
  const int wave = (blockIdx.x * blockDim.x + threadIdx.x) >> 5;
  const int nwaves = (gridDim.x * blockDim.x) >> 5;
  const int tilesM = M >> 4, tilesN4 = N >> 6;
  const int l = lane & 15, hi = lane >> 4;
  for (int t = wave; t < tilesM * tilesN4; t += nwaves) {
    const int m0 = (t / tilesN4) << 4;
    const int n0 = (t % tilesN4) << 6;
    v8f acc[4];
#pragma unroll
    for (int j = 0; j < 4; ++j) {
      const float bv = bias[n0 + j * 16 + l];
#pragma unroll
      for (int r = 0; r < 8; ++r) acc[j][r] = bv;
    }
    const bf16* pa = A + (size_t)(m0 + l) * lda + hi * 16;
    const bf16* pb = W + (size_t)(n0 + l) * ldw + hi * 16;
    const size_t wstep = (size_t)ldw * 16;
    for (int k = 0; k < KC; k += 32) {
      v8bf a0 = *(const v8bf*)(pa + k);
      v8bf a1 = *(const v8bf*)(pa + k + 8);
      __builtin_prefetch(pa + k + 256, 0, 1);
      v16bf a = pack16(a0, a1);
#pragma unroll
      for (int j = 0; j < 4; ++j) {
        const bf16* pbj = pb + (size_t)j * wstep + k;
        v8bf b0 = *(const v8bf*)pbj;
        v8bf b1 = *(const v8bf*)(pbj + 8);
        v16bf b = pack16(b0, b1);
        acc[j] = __builtin_amdgcn_wmma_f32_16x16x32_bf16(false, a, false, b,
                                                         (short)0, acc[j],
                                                         false, false);
      }
    }
#pragma unroll
    for (int j = 0; j < 4; ++j) {
#pragma unroll
      for (int r = 0; r < 8; ++r) {
        const int row = m0 + hi * 8 + r, col = n0 + j * 16 + l;
        size_t o;
        if (permute) { const int bb = row % PB, tt = row / PB;
                       o = ((size_t)bb * PT + tt) * N + col; }
        else         { o = (size_t)row * N + col; }
        C[o] = acc[j][r];
      }
    }
  }
}

// ---------------------------------------------------------------------------
// Stage 3: persistent GRU-ODE recurrence. 64 blocks x 256 thr (512 waves).
// Whh slice (one M-tile x 8 N-tiles, 128x256 bf16) DMA'd to LDS once by the
// TDM (row pad 16B -> conflict-free reads), reused for all 255 steps.
// Per step: [gemm1 tanh] sync [gemm2 h_ode] sync [gemm3 gh] sync [gates] sync
// ---------------------------------------------------------------------------
__global__ __launch_bounds__(256) void gru_ode_kernel(
    const float* __restrict__ ts, const float* __restrict__ gi_all,
    const bf16* __restrict__ Wo1p, const float* __restrict__ bo1,
    const bf16* __restrict__ Wo2b, const float* __restrict__ bo2,
    const bf16* __restrict__ Whhb, const float* __restrict__ bhh,
    const float* __restrict__ s_ptr,
    float* __restrict__ hF, bf16* __restrict__ hab,
    bf16* __restrict__ aob, float* __restrict__ hodeF, bf16* __restrict__ hodeB,
    float* __restrict__ ghBuf, float* __restrict__ dtv,
    bf16* __restrict__ latB, unsigned* bar) {
  extern __shared__ __align__(16) char dynsmem[];
  bf16* ldsW = (bf16*)dynsmem;               // 128 rows x PITCH_GRU = 67584B

  const int tid = blockIdx.x * blockDim.x + threadIdx.x;
  const int nthreads = gridDim.x * blockDim.x;
  const int lane = threadIdx.x & 31;
  const int wave = tid >> 5;
  const int nwaves = nthreads >> 5;
  const int l = lane & 15, hi = lane >> 4;
  const float abs_s = fabsf(s_ptr[0]);

  // Stage this block's Whh slice (rows [(blk%6)*128, +128)) into LDS once.
  // 512B rows -> pad_interval code 6 ((1<<6)*8B == 512B).
  if (blockIdx.x < 48)
    stage_slice(Whhb + (size_t)(((int)blockIdx.x % 6) * 128) * Z_,
                ldsW, 128, Z_, PITCH_GRU, 6);

  // ---- init: zero padded aug buffer; h1 = sgru(0, 0, enh[:,0]) (gh == bhh)
  for (int i = tid; i < B_ * KP_; i += nthreads) hab[i] = f2bf(0.0f);
  for (int i = tid; i < B_ * Z_; i += nthreads) {
    const int b = i / Z_, z = i % Z_;
    const float* gi = gi_all + ((size_t)b * T_) * G3_;
    const float r  = sigm(gi[z] + bhh[z]);
    const float zg = sigm(gi[Z_ + z] + bhh[Z_ + z]);
    const float n  = tanhf(gi[2 * Z_ + z] + r * bhh[2 * Z_ + z]);
    const float ht = (1.0f - zg) * n;                 // h_ode == 0
    const float hn = (fabsf(ht) > 0.0f) ? ht : 0.0f;  // skip vs h_prev == 0
    hF[i] = hn;
    hab[b * KP_ + z] = f2bf(hn);
    latB[((size_t)b) * Z_ + z] = f2bf(hn);            // latent row t=0: 0*B+b
  }
  for (int b = tid; b < B_; b += nthreads) hab[b * KP_ + Z_] = f2bf(ts[b * T_]);
  __syncthreads();                                    // TDM data visible in WG
  grid_sync(bar);

  for (int t = 1; t < T_; ++t) {
    // dt for this step (consumed after next sync)
    for (int b = tid; b < B_; b += nthreads)
      dtv[b] = fmaxf(fabsf(ts[b * T_ + t] - ts[b * T_ + t - 1]), 1e-6f);

    // phase 1: ao = tanh([h,pt] @ Wo1^T + bo1)  (128x288 x 288x128)
    for (int tt = wave; tt < (B_ / 16) * (U_ / 16); tt += nwaves) {
      const int m0 = (tt / (U_ / 16)) << 4, n0 = (tt % (U_ / 16)) << 4;
      v8f acc; const float bv = bo1[n0 + l];
#pragma unroll
      for (int r = 0; r < 8; ++r) acc[r] = bv;
      acc = wmma_tile(hab, KP_, Wo1p, KP_, m0, n0, KP_, lane, acc);
#pragma unroll
      for (int r = 0; r < 8; ++r)
        aob[(m0 + hi * 8 + r) * U_ + n0 + l] = f2bf(tanhf(acc[r]));
    }
    grid_sync(bar);

    // phase 2: h_ode = h + (ao @ Wo2^T + bo2) * dt  (128x128 x 128x256)
    for (int tt = wave; tt < (B_ / 16) * (Z_ / 16); tt += nwaves) {
      const int m0 = (tt / (Z_ / 16)) << 4, n0 = (tt % (Z_ / 16)) << 4;
      v8f acc; const float bv = bo2[n0 + l];
#pragma unroll
      for (int r = 0; r < 8; ++r) acc[r] = bv;
      acc = wmma_tile(aob, U_, Wo2b, U_, m0, n0, U_, lane, acc);
#pragma unroll
      for (int r = 0; r < 8; ++r) {
        const int row = m0 + hi * 8 + r, col = n0 + l;
        const float ho = hF[row * Z_ + col] + acc[r] * dtv[row];
        hodeF[row * Z_ + col] = ho;
        hodeB[row * Z_ + col] = f2bf(ho);
      }
    }
    grid_sync(bar);

    // phase 3: gh = h_ode @ Whh^T + bhh  (128x256 x 256x768), B from LDS.
    // Static map: block blk (<48) owns m-tile blk/6, n-tiles (blk%6)*8+wblk.
    if (blockIdx.x < 48) {
      const int wblk = threadIdx.x >> 5;
      const int m0 = ((int)blockIdx.x / 6) << 4;
      const int n0 = (((int)blockIdx.x % 6) * 8 + wblk) << 4;
      v8f acc; const float bv = bhh[n0 + l];
#pragma unroll
      for (int r = 0; r < 8; ++r) acc[r] = bv;
      acc = wmma_tile_ldsB(hodeB, Z_, ldsW, PITCH_GRU, wblk * 16, m0, Z_, lane, acc);
#pragma unroll
      for (int r = 0; r < 8; ++r)
        ghBuf[(m0 + hi * 8 + r) * G3_ + n0 + l] = acc[r];
    }
    grid_sync(bar);

    // phase 4: gates + skip-update + emit latent
    for (int i = tid; i < B_ * Z_; i += nthreads) {
      const int b = i / Z_, z = i % Z_;
      const float* gi = gi_all + ((size_t)b * T_ + t) * G3_;
      const float* g  = ghBuf + b * G3_;
      const float r  = sigm(gi[z] + g[z]);
      const float zg = sigm(gi[Z_ + z] + g[Z_ + z]);
      const float n  = tanhf(gi[2 * Z_ + z] + r * g[2 * Z_ + z]);
      const float ho = hodeF[i];
      const float ht = (1.0f - zg) * n + zg * ho;
      const float hp = hF[i];
      const float hn = (fabsf(ht - hp) > fabsf(hp) * abs_s) ? ht : hp;
      hF[i] = hn;
      hab[b * KP_ + z] = f2bf(hn);
      latB[((size_t)t * B_ + b) * Z_ + z] = f2bf(hn);
    }
    for (int b = tid; b < B_; b += nthreads)
      hab[b * KP_ + Z_] = f2bf(ts[b * T_ + t]);       // pt <- ct
    grid_sync(bar);
  }
}

// ---------------------------------------------------------------------------
// Stage 4: bidirectional GRU; fwd on blocks [0,32), bwd on [32,64).
// Per-direction W_hh slice (128x128 bf16) TDM-staged to LDS once.
// ---------------------------------------------------------------------------
__global__ __launch_bounds__(256) void bigru_kernel(
    const float* __restrict__ giF, const float* __restrict__ giB,
    const bf16* __restrict__ Wfhh, const float* __restrict__ bfhh,
    const bf16* __restrict__ Wbhh, const float* __restrict__ bbhh,
    float* __restrict__ hfF, bf16* __restrict__ hfB,
    float* __restrict__ hbF, bf16* __restrict__ hbB,
    float* __restrict__ ghf, float* __restrict__ ghb,
    bf16* __restrict__ catB, unsigned* bar) {
  extern __shared__ __align__(16) char dynsmem[];
  bf16* ldsW = (bf16*)dynsmem;               // 128 rows x PITCH_BIGRU = 34816B

  const int halfBlocks = gridDim.x / 2;
  const int dir = (blockIdx.x >= (unsigned)halfBlocks) ? 1 : 0;
  const int blk = blockIdx.x - dir * halfBlocks;
  const int dtid = blk * blockDim.x + threadIdx.x;
  const int dthreads = halfBlocks * blockDim.x;
  const int lane = threadIdx.x & 31;
  const int l = lane & 15, hi = lane >> 4;

  const float* giA = dir ? giB : giF;
  const bf16*  Wh  = dir ? Wbhh : Wfhh;
  const float* bh  = dir ? bbhh : bfhh;
  float* hD  = dir ? hbF : hfF;
  bf16*  hDb = dir ? hbB : hfB;
  float* ghD = dir ? ghb : ghf;

  // Stage this block's W_hh slice (rows [(blk%3)*128, +128)) into LDS once.
  // 256B rows -> pad_interval code 5 ((1<<5)*8B == 256B).
  if (blk < 24)
    stage_slice(Wh + (size_t)((blk % 3) * 128) * H_, ldsW, 128, H_,
                PITCH_BIGRU, 5);

  for (int i = dtid; i < B_ * H_; i += dthreads) { hD[i] = 0.0f; hDb[i] = f2bf(0.0f); }
  __syncthreads();                                    // TDM data visible in WG
  grid_sync(bar);

  for (int step = 0; step < T_; ++step) {
    const int t = dir ? (T_ - 1 - step) : step;
    // gh = h @ W_hh^T + b_hh  (128x128 x 128x384), B from LDS.
    // Static map: block blk (<24) owns m-tile blk/3, n-tiles (blk%3)*8+wblk.
    if (blk < 24) {
      const int wblk = threadIdx.x >> 5;
      const int m0 = (blk / 3) << 4;
      const int n0 = ((blk % 3) * 8 + wblk) << 4;
      v8f acc; const float bv = bh[n0 + l];
#pragma unroll
      for (int r = 0; r < 8; ++r) acc[r] = bv;
      acc = wmma_tile_ldsB(hDb, H_, ldsW, PITCH_BIGRU, wblk * 16, m0, H_, lane, acc);
#pragma unroll
      for (int r = 0; r < 8; ++r)
        ghD[(m0 + hi * 8 + r) * GH_ + n0 + l] = acc[r];
    }
    grid_sync(bar);
    // gates
    for (int i = dtid; i < B_ * H_; i += dthreads) {
      const int b = i / H_, z = i % H_;
      const float* gi = giA + ((size_t)t * B_ + b) * GH_;
      const float* g  = ghD + b * GH_;
      const float r  = sigm(gi[z] + g[z]);
      const float zg = sigm(gi[H_ + z] + g[H_ + z]);
      const float n  = tanhf(gi[2 * H_ + z] + r * g[2 * H_ + z]);
      const float hn = (1.0f - zg) * n + zg * hD[i];
      hD[i] = hn;
      hDb[i] = f2bf(hn);
      catB[((size_t)t * B_ + b) * Z_ + dir * H_ + z] = f2bf(hn);
    }
    grid_sync(bar);
  }
}

// ---------------------------------------------------------------------------
extern "C" void kernel_launch(void* const* d_in, const int* in_sizes, int n_in,
                              void* d_out, int out_size, void* d_ws, size_t ws_size,
                              hipStream_t stream) {
  (void)in_sizes; (void)n_in; (void)out_size; (void)ws_size;
  const float* x    = (const float*)d_in[0];
  const float* ts   = (const float*)d_in[1];
  const float* Wc1  = (const float*)d_in[2];
  const float* bc1  = (const float*)d_in[3];
  const float* Wc2  = (const float*)d_in[4];
  const float* bc2  = (const float*)d_in[5];
  const float* Wc3  = (const float*)d_in[6];
  const float* bc3  = (const float*)d_in[7];
  const float* Wo1  = (const float*)d_in[8];
  const float* bo1  = (const float*)d_in[9];
  const float* Wo2  = (const float*)d_in[10];
  const float* bo2  = (const float*)d_in[11];
  const float* Wih  = (const float*)d_in[12];
  const float* Whh  = (const float*)d_in[13];
  const float* bih  = (const float*)d_in[14];
  const float* bhh  = (const float*)d_in[15];
  const float* s    = (const float*)d_in[16];
  const float* Wfih = (const float*)d_in[17];
  const float* Wfhh = (const float*)d_in[18];
  const float* bfih = (const float*)d_in[19];
  const float* bfhh = (const float*)d_in[20];
  const float* Wbih = (const float*)d_in[21];
  const float* Wbhh = (const float*)d_in[22];
  const float* bbih = (const float*)d_in[23];
  const float* bbhh = (const float*)d_in[24];
  const float* Wp   = (const float*)d_in[25];
  const float* bp   = (const float*)d_in[26];

  char* p = (char*)d_ws;
  auto bump = [&](size_t bytes) -> void* {
    void* r = (void*)p; p += (bytes + 255) & ~(size_t)255; return r;
  };
  const int MT = B_ * T_;                                  // 32768
  bf16*  enh_b  = (bf16*)  bump((size_t)MT * E_ * 2);      //  16.8 MB
  float* gi     = (float*) bump((size_t)MT * G3_ * 4);     // 100.7 MB
  bf16*  latB   = (bf16*)  bump((size_t)MT * Z_ * 2);      //  16.8 MB
  float* giF    = (float*) bump((size_t)MT * GH_ * 4);     //  50.3 MB
  float* giBw   = (float*) bump((size_t)MT * GH_ * 4);     //  50.3 MB
  bf16*  catB   = (bf16*)  bump((size_t)MT * Z_ * 2);      //  16.8 MB
  bf16*  Wih_b  = (bf16*)  bump((size_t)G3_ * E_ * 2);
  bf16*  Whh_b  = (bf16*)  bump((size_t)G3_ * Z_ * 2);
  bf16*  Wo1p   = (bf16*)  bump((size_t)U_ * KP_ * 2);
  bf16*  Wo2_b  = (bf16*)  bump((size_t)Z_ * U_ * 2);
  bf16*  Wfih_b = (bf16*)  bump((size_t)GH_ * Z_ * 2);
  bf16*  Wfhh_b = (bf16*)  bump((size_t)GH_ * H_ * 2);
  bf16*  Wbih_b = (bf16*)  bump((size_t)GH_ * Z_ * 2);
  bf16*  Wbhh_b = (bf16*)  bump((size_t)GH_ * H_ * 2);
  bf16*  Wp_b   = (bf16*)  bump((size_t)Z_ * Z_ * 2);
  float* hF     = (float*) bump((size_t)B_ * Z_ * 4);
  bf16*  hab    = (bf16*)  bump((size_t)B_ * KP_ * 2);
  bf16*  aob    = (bf16*)  bump((size_t)B_ * U_ * 2);
  float* hodeF  = (float*) bump((size_t)B_ * Z_ * 4);
  bf16*  hodeB  = (bf16*)  bump((size_t)B_ * Z_ * 2);
  float* ghBuf  = (float*) bump((size_t)B_ * G3_ * 4);
  float* dtv    = (float*) bump((size_t)B_ * 4);
  float* hfF    = (float*) bump((size_t)B_ * H_ * 4);
  bf16*  hfB    = (bf16*)  bump((size_t)B_ * H_ * 2);
  float* hbF    = (float*) bump((size_t)B_ * H_ * 4);
  bf16*  hbB    = (bf16*)  bump((size_t)B_ * H_ * 2);
  float* ghf    = (float*) bump((size_t)B_ * GH_ * 4);
  float* ghb    = (float*) bump((size_t)B_ * GH_ * 4);
  unsigned* bar = (unsigned*)bump(256);

  (void)hipMemsetAsync(bar, 0, 256, stream);

  auto cvt = [&](const float* src, bf16* dst, int n) {
    cvt_bf16<<<(n + 255) / 256, 256, 0, stream>>>(src, dst, n);
  };
  cvt(Wih,  Wih_b,  G3_ * E_);
  cvt(Whh,  Whh_b,  G3_ * Z_);
  cvt(Wo2,  Wo2_b,  Z_ * U_);
  cvt(Wfih, Wfih_b, GH_ * Z_);
  cvt(Wfhh, Wfhh_b, GH_ * H_);
  cvt(Wbih, Wbih_b, GH_ * Z_);
  cvt(Wbhh, Wbhh_b, GH_ * H_);
  cvt(Wp,   Wp_b,   Z_ * Z_);
  pad_wo1<<<(U_ * KP_ + 255) / 256, 256, 0, stream>>>(Wo1, Wo1p);

  // Stage 1: temporal conv -> enh (bf16)
  tconv_kernel<<<MT, 256, 0, stream>>>(x, ts, Wc1, bc1, Wc2, bc2, Wc3, bc3, enh_b);

  // Stage 2: hoisted input GEMM  gi = enh @ Wih^T + bih  (32768x768, K=256)
  gemm_bias_kernel<E_><<<2048, 256, 0, stream>>>(enh_b, E_, Wih_b, E_, bih, gi,
                                                 MT, G3_, 0, 0, 0);

  // Stage 3: persistent GRU-ODE recurrence (67584B LDS: padded Whh slice)
  gru_ode_kernel<<<64, 256, 128 * PITCH_GRU * 2, stream>>>(
      ts, gi, Wo1p, bo1, Wo2_b, bo2, Whh_b, bhh, s, hF, hab, aob,
      hodeF, hodeB, ghBuf, dtv, latB, bar);

  // Stage 4a: hoisted BiGRU input GEMMs (32768x384, K=256)
  gemm_bias_kernel<Z_><<<2048, 256, 0, stream>>>(latB, Z_, Wfih_b, Z_, bfih, giF,
                                                 MT, GH_, 0, 0, 0);
  gemm_bias_kernel<Z_><<<2048, 256, 0, stream>>>(latB, Z_, Wbih_b, Z_, bbih, giBw,
                                                 MT, GH_, 0, 0, 0);

  // Stage 4b: fwd+bwd recurrences concurrently (34816B LDS: padded W_hh slice)
  bigru_kernel<<<64, 256, 128 * PITCH_BIGRU * 2, stream>>>(
      giF, giBw, Wfhh_b, bfhh, Wbhh_b, bbhh,
      hfF, hfB, hbF, hbB, ghf, ghb, catB, bar);

  // Stage 5: projection + (T,B)->(B,T) permute straight into d_out
  gemm_bias_kernel<Z_><<<2048, 256, 0, stream>>>(catB, Z_, Wp_b, Z_, bp,
                                                 (float*)d_out, MT, Z_,
                                                 1, B_, T_);
}
